// PaperCompliantResponseDecoder_23940147707906
// MI455X (gfx1250) — matrix operand order repaired and checked by reading, not compiled, not run on hardware
//
#include <hip/hip_runtime.h>
#include <hip/hip_bf16.h>
#include <stdint.h>

// ---------------- problem constants ----------------
#define BB   8
#define TT   128
#define LEE  128
#define DD   512
#define HH   8
#define DH   64          // DD / HH
#define DFFN 2048
#define VV   32000
#define VEXT 32100
#define BT   (BB * TT)           // 1024
#define BTD  (BB * TT * DD)      // 524288

typedef __attribute__((ext_vector_type(16))) __bf16 bfx16;
typedef __attribute__((ext_vector_type(8)))  float  fx8;

// gfx1250 async global->LDS copy (ASYNCcnt-tracked); inline asm per CDNA5 doc.
__device__ __forceinline__ void async_copy_b128(uint32_t lds_off, const void* gaddr) {
    asm volatile("global_load_async_to_lds_b128 %0, %1, off"
                 :: "v"(lds_off), "v"(gaddr) : "memory");
}
#define WAIT_ASYNC_LE(n) asm volatile("s_wait_asynccnt " #n ::: "memory")

// ---------------- block reduce helpers ----------------
__device__ __forceinline__ float block_reduce_sum(float v, float* red) {
    int tid = threadIdx.x;
    red[tid] = v;
    __syncthreads();
    for (int s = blockDim.x >> 1; s > 0; s >>= 1) {
        if (tid < s) red[tid] += red[tid + s];
        __syncthreads();
    }
    float r = red[0];
    __syncthreads();
    return r;
}

__device__ __forceinline__ float block_reduce_max(float v, float* red) {
    int tid = threadIdx.x;
    red[tid] = v;
    __syncthreads();
    for (int s = blockDim.x >> 1; s > 0; s >>= 1) {
        if (tid < s) red[tid] = fmaxf(red[tid], red[tid + s]);
        __syncthreads();
    }
    float r = red[0];
    __syncthreads();
    return r;
}

// ---------------- fp32 -> bf16 conversion (weights, one pass) ----------------
__global__ void cvt_bf16_kernel(const float* __restrict__ src,
                                __bf16* __restrict__ dst, long long n) {
    long long i = (long long)blockIdx.x * blockDim.x + threadIdx.x;
    if (i < n) dst[i] = (__bf16)src[i];
}

// ---------------- embedding + positional add ----------------
__global__ void embed_kernel(const int* __restrict__ ids,
                             const float* __restrict__ tok,
                             const float* __restrict__ pos,
                             float* __restrict__ emb,
                             float* __restrict__ x) {
    long long i = (long long)blockIdx.x * blockDim.x + threadIdx.x;
    if (i >= (long long)BTD) return;
    int d  = (int)(i % DD);
    int bt = (int)(i / DD);
    int t  = bt % TT;
    float e = tok[(long long)ids[bt] * DD + d];
    emb[i] = e;
    x[i]   = e + pos[(long long)t * DD + d];
}

// ---------------- layernorm: fp32 in, bf16 out (feeds WMMA GEMMs) ----------------
__launch_bounds__(128)
__global__ void ln_kernel(const float* __restrict__ x,
                          const float* __restrict__ g,
                          const float* __restrict__ b,
                          __bf16* __restrict__ out) {
    __shared__ float red[128];
    int row = blockIdx.x;
    int tid = threadIdx.x;
    const float* xr = x + (long long)row * DD;
    float v[4];
    float s = 0.f;
    #pragma unroll
    for (int i = 0; i < 4; ++i) { v[i] = xr[tid + i * 128]; s += v[i]; }
    float mean = block_reduce_sum(s, red) * (1.0f / DD);
    float sq = 0.f;
    #pragma unroll
    for (int i = 0; i < 4; ++i) { float d = v[i] - mean; sq += d * d; }
    float var = block_reduce_sum(sq, red) * (1.0f / DD);
    float inv = rsqrtf(var + 1e-5f);
    __bf16* orow = out + (long long)row * DD;
    #pragma unroll
    for (int i = 0; i < 4; ++i) {
        int c = tid + i * 128;
        orow[c] = (__bf16)((v[i] - mean) * inv * g[c] + b[c]);
    }
}

// ---------------- bf16 WMMA GEMM with async-LDS double buffering ----------------
// C = act( A[M,K](bf16) @ B + bias ) (+ residual)
//   b_is_kn == 0 : B is W[N,K] bf16 row-major (A @ W^T), tiles staged with
//                  GLOBAL_LOAD_ASYNC_TO_LDS_B128 (ASYNCcnt double buffering)
//   b_is_kn == 1 : B is [K,N] bf16 row-major (A @ B), B staged via DS stores
// Block tile 64x64, BK=32, 128 threads (4 waves); wave w owns rows [16w,16w+16).
#define GBM 64
#define GBN 64
#define GBK 32
#define LSTR 40   // padded bf16 row stride in LDS (80B rows, 16B-aligned chunks)

__launch_bounds__(128)
__global__ void gemm_wmma_bf16(const __bf16* __restrict__ A,
                               const __bf16* __restrict__ Bm,
                               const float* __restrict__ bias,
                               const float* __restrict__ resid,
                               float* __restrict__ Cf,
                               __bf16* __restrict__ Cb,
                               int M, int N, int K,
                               int lda, int ldb, int ldc,
                               long long strideA, long long strideB,
                               long long strideC, long long strideR,
                               int b_is_kn, int relu_flag) {
    __shared__ __align__(16) __bf16 As[2][GBM][LSTR];
    __shared__ __align__(16) __bf16 Bs[2][GBN][LSTR];

    int tid  = threadIdx.x;
    int lane = tid & 31;
    int wave = tid >> 5;
    int r    = lane & 15;
    int half = lane >> 4;
    int mr   = wave * 16;

    long long bz = blockIdx.z;
    const __bf16* Ab = A  + bz * strideA;
    const __bf16* Bb = Bm + bz * strideB;

    int m0 = blockIdx.y * GBM;
    int n0 = blockIdx.x * GBN;

    fx8 acc[4] = {};

    auto stage = [&](int kt, int buf) {
        int k0 = kt * GBK;
        // A tile: 64 rows x 64B, 256 x 16B chunks, async DMA to LDS
        #pragma unroll
        for (int i = 0; i < 2; ++i) {
            int c   = tid * 2 + i;
            int row = c >> 2;
            int ch  = c & 3;
            const __bf16* ga = Ab + (long long)(m0 + row) * lda + k0 + ch * 8;
            async_copy_b128((uint32_t)(uintptr_t)&As[buf][row][ch * 8], ga);
        }
        if (!b_is_kn) {
            #pragma unroll
            for (int i = 0; i < 2; ++i) {
                int c   = tid * 2 + i;
                int row = c >> 2;      // n within tile
                int ch  = c & 3;
                const __bf16* gb = Bb + (long long)(n0 + row) * ldb + k0 + ch * 8;
                async_copy_b128((uint32_t)(uintptr_t)&Bs[buf][row][ch * 8], gb);
            }
        } else {
            // transpose gather: Bs[n][k] = B[k][n]
            #pragma unroll
            for (int i = 0; i < 16; ++i) {
                int idx = tid * 16 + i;
                int n = idx & 63;
                int k = idx >> 6;
                Bs[buf][n][k] = Bb[(long long)(k0 + k) * ldb + n0 + n];
            }
        }
    };

    int nk = K / GBK;
    stage(0, 0);
    for (int kt = 0; kt < nk; ++kt) {
        int cur = kt & 1;
        if (kt + 1 < nk) {
            stage(kt + 1, cur ^ 1);
            // in-order completion: <=4 (or <=2) outstanding => current tile done
            if (b_is_kn) { WAIT_ASYNC_LE(2); } else { WAIT_ASYNC_LE(4); }
        } else {
            WAIT_ASYNC_LE(0);
        }
        __syncthreads();

        // A fragment (16x32 bf16, ISA lane layout)
        bfx16 af;
        {
            const __bf16* rp = &As[cur][mr + r][0];
            int kb = half * 8;
            #pragma unroll
            for (int e = 0; e < 8; ++e) af[e] = rp[kb + e];
            #pragma unroll
            for (int e = 0; e < 8; ++e) af[8 + e] = rp[16 + kb + e];
        }
        // all 4 B fragments first, then back-to-back WMMAs (A reuse hints)
        bfx16 bfv[4];
        #pragma unroll
        for (int j = 0; j < 4; ++j) {
            const __bf16* rp = &Bs[cur][j * 16 + r][0];
            int kb = half * 8;
            #pragma unroll
            for (int e = 0; e < 8; ++e) bfv[j][e] = rp[kb + e];
            #pragma unroll
            for (int e = 0; e < 8; ++e) bfv[j][8 + e] = rp[16 + kb + e];
        }
        acc[0] = __builtin_amdgcn_wmma_f32_16x16x32_bf16(false, af, false, bfv[0], (short)0, acc[0], true,  false);
        acc[1] = __builtin_amdgcn_wmma_f32_16x16x32_bf16(false, af, false, bfv[1], (short)0, acc[1], true,  false);
        acc[2] = __builtin_amdgcn_wmma_f32_16x16x32_bf16(false, af, false, bfv[2], (short)0, acc[2], true,  false);
        acc[3] = __builtin_amdgcn_wmma_f32_16x16x32_bf16(false, af, false, bfv[3], (short)0, acc[3], false, false);
        __syncthreads();
    }

    // epilogue: bias (+relu) (+residual), fp32 and/or bf16 outputs
    float*  Cfb = Cf ? (Cf + bz * strideC) : nullptr;
    __bf16* Cbb = Cb ? (Cb + bz * strideC) : nullptr;
    const float* Rb = resid ? (resid + bz * strideR) : nullptr;
    #pragma unroll
    for (int j = 0; j < 4; ++j) {
        int col = n0 + j * 16 + r;
        float bv = bias ? bias[col] : 0.f;
        #pragma unroll
        for (int i = 0; i < 8; ++i) {
            int row = m0 + mr + half * 8 + i;
            float v = acc[j][i] + bv;
            if (relu_flag) v = v > 0.f ? v : 0.f;
            long long off = (long long)row * ldc + col;
            if (Rb)  v += Rb[off];
            if (Cfb) Cfb[off] = v;
            if (Cbb) Cbb[off] = (__bf16)v;
        }
    }
}

// ---------------- self-attention (causal), online softmax, bf16 I/O ----------------
__launch_bounds__(128)
__global__ void self_attn_kernel(const __bf16* __restrict__ qkv,  // (B*T, 1536) bf16
                                 __bf16* __restrict__ out) {      // (B*T, 512) bf16
    __shared__ float Ks[TT][DH];
    __shared__ float Vs[TT][DH];
    int bh = blockIdx.x;
    int b = bh / HH, h = bh % HH;
    int tid = threadIdx.x;
    const __bf16* base = qkv + (long long)(b * TT) * (3 * DD);
    #pragma unroll 4
    for (int d = 0; d < DH; ++d) {
        Ks[tid][d] = (float)base[(long long)tid * (3 * DD) + DD     + h * DH + d];
        Vs[tid][d] = (float)base[(long long)tid * (3 * DD) + 2 * DD + h * DH + d];
    }
    __syncthreads();

    float q[DH];
    #pragma unroll 4
    for (int d = 0; d < DH; ++d) q[d] = (float)base[(long long)tid * (3 * DD) + h * DH + d];

    float m = -1e30f, l = 0.f;
    float o[DH];
    #pragma unroll 4
    for (int d = 0; d < DH; ++d) o[d] = 0.f;

    for (int k = 0; k < TT; ++k) {
        float x = 0.f;
        #pragma unroll 4
        for (int d = 0; d < DH; ++d) x += q[d] * Ks[k][d];
        x *= 0.125f;                       // 1/sqrt(64)
        if (k > tid) x += -1e9f;           // causal mask
        float mn = fmaxf(m, x);
        float sc = __expf(m - mn);
        float e  = __expf(x - mn);
        l = l * sc + e;
        #pragma unroll 4
        for (int d = 0; d < DH; ++d) o[d] = o[d] * sc + e * Vs[k][d];
        m = mn;
    }
    float inv = 1.f / l;
    __bf16* orow = out + (long long)(b * TT + tid) * DD + h * DH;
    #pragma unroll 4
    for (int d = 0; d < DH; ++d) orow[d] = (__bf16)(o[d] * inv);
}

// ---------------- cross-attention: probs (fp32) + context (bf16) ----------------
__launch_bounds__(128)
__global__ void cross_attn_kernel(const __bf16* __restrict__ qc,   // (B*T, 512)
                                  const __bf16* __restrict__ kvc,  // (B*LE, 1024)
                                  float* __restrict__ probs,       // (B,H,T,LE)
                                  __bf16* __restrict__ out) {      // (B*T, 512)
    __shared__ float Ks[LEE][DH];
    __shared__ float Vs[LEE][DH];
    int bh = blockIdx.x;
    int b = bh / HH, h = bh % HH;
    int tid = threadIdx.x;
    const __bf16* kvbase = kvc + (long long)(b * LEE) * (2 * DD);
    #pragma unroll 4
    for (int d = 0; d < DH; ++d) {
        Ks[tid][d] = (float)kvbase[(long long)tid * (2 * DD)      + h * DH + d];
        Vs[tid][d] = (float)kvbase[(long long)tid * (2 * DD) + DD + h * DH + d];
    }
    __syncthreads();

    float q[DH];
    const __bf16* qrow = qc + (long long)(b * TT + tid) * DD + h * DH;
    #pragma unroll 4
    for (int d = 0; d < DH; ++d) q[d] = (float)qrow[d];

    // pass 1: running max / sum
    float m = -1e30f, l = 0.f;
    for (int k = 0; k < LEE; ++k) {
        float x = 0.f;
        #pragma unroll 4
        for (int d = 0; d < DH; ++d) x += q[d] * Ks[k][d];
        x *= 0.125f;
        float mn = fmaxf(m, x);
        l = l * __expf(m - mn) + __expf(x - mn);
        m = mn;
    }
    float inv = 1.f / l;

    // pass 2: probabilities + context
    float o[DH];
    #pragma unroll 4
    for (int d = 0; d < DH; ++d) o[d] = 0.f;
    float* prow = probs + (((long long)(b * HH + h) * TT + tid) * LEE);
    for (int k = 0; k < LEE; ++k) {
        float x = 0.f;
        #pragma unroll 4
        for (int d = 0; d < DH; ++d) x += q[d] * Ks[k][d];
        x *= 0.125f;
        float p = __expf(x - m) * inv;
        prow[k] = p;
        #pragma unroll 4
        for (int d = 0; d < DH; ++d) o[d] += p * Vs[k][d];
    }
    __bf16* orow = out + (long long)(b * TT + tid) * DD + h * DH;
    #pragma unroll 4
    for (int d = 0; d < DH; ++d) orow[d] = (__bf16)o[d];
}

// ---------------- average attention over heads (fp32 out + bf16 copy) ----------------
__global__ void avg_probs_kernel(const float* __restrict__ probs,   // (B,H,T,LE)
                                 float* __restrict__ attn_avg,      // (B,T,LE) fp32
                                 __bf16* __restrict__ attn_avg_bf) {// (B,T,LE) bf16
    long long i = (long long)blockIdx.x * blockDim.x + threadIdx.x;
    if (i >= (long long)BB * TT * LEE) return;
    int l = (int)(i % LEE);
    int t = (int)((i / LEE) % TT);
    int b = (int)(i / ((long long)TT * LEE));
    float s = 0.f;
    #pragma unroll
    for (int h = 0; h < HH; ++h)
        s += probs[(((long long)(b * HH + h) * TT + t) * LEE) + l];
    float a = s * (1.0f / HH);
    attn_avg[i] = a;
    attn_avg_bf[i] = (__bf16)a;
}

// ---------------- softmax over extended vocab row ----------------
__launch_bounds__(256)
__global__ void softmax_ext_kernel(float* __restrict__ P) {  // rows x VEXT, logits in [0,V)
    __shared__ float red[256];
    int row = blockIdx.x;
    int tid = threadIdx.x;
    float* pr = P + (long long)row * VEXT;
    float m = -1e30f;
    for (int i = tid; i < VV; i += 256) m = fmaxf(m, pr[i]);
    m = block_reduce_max(m, red);
    float s = 0.f;
    for (int i = tid; i < VV; i += 256) {
        float e = __expf(pr[i] - m);
        pr[i] = e;
        s += e;
    }
    s = block_reduce_sum(s, red);
    float inv = 1.f / s;
    for (int i = tid; i < VV; i += 256) pr[i] *= inv;
    for (int i = VV + tid; i < VEXT; i += 256) pr[i] = 0.f;
}

// ---------------- pointer-generator gate ----------------
__launch_bounds__(128)
__global__ void pmix_kernel(const float* __restrict__ hidden,  // (BT, D)
                            const float* __restrict__ ctx,     // (BT, D)
                            const float* __restrict__ emb,     // (BT, D)
                            const float* __restrict__ pg_w,    // (1, 3D)
                            const float* __restrict__ pg_b,    // (1,)
                            float* __restrict__ pmix) {        // (BT,)
    __shared__ float red[128];
    int row = blockIdx.x;
    int tid = threadIdx.x;
    long long base = (long long)row * DD;
    float s = 0.f;
    for (int i = tid; i < DD; i += 128) {
        s += hidden[base + i] * pg_w[i]
           + ctx[base + i]    * pg_w[DD + i]
           + emb[base + i]    * pg_w[2 * DD + i];
    }
    s = block_reduce_sum(s, red);
    if (tid == 0) pmix[row] = 1.f / (1.f + __expf(-(s + pg_b[0])));
}

// ---------------- zero fill ----------------
__global__ void zero_kernel(float* __restrict__ p, long long n) {
    long long i = (long long)blockIdx.x * blockDim.x + threadIdx.x;
    if (i < n) p[i] = 0.f;
}

// ---------------- Pcopy scatter-add ----------------
__global__ void pcopy_scatter_kernel(const float* __restrict__ attn_avg,  // (B,T,LE)
                                     const int* __restrict__ src_ids,    // (B,LE)
                                     float* __restrict__ Pcopy) {        // (B,T,VEXT)
    long long i = (long long)blockIdx.x * blockDim.x + threadIdx.x;
    if (i >= (long long)BB * TT * LEE) return;
    int l = (int)(i % LEE);
    long long bt = i / LEE;
    int b = (int)(bt / TT);
    int v = src_ids[(long long)b * LEE + l];
    atomicAdd(&Pcopy[bt * VEXT + v], attn_avg[i]);
}

// ---------------- final mix ----------------
__global__ void pw_kernel(const float* __restrict__ pmix,
                          const float* __restrict__ Pgen,
                          const float* __restrict__ Pcopy,
                          float* __restrict__ Pw, long long n) {
    long long i = (long long)blockIdx.x * blockDim.x + threadIdx.x;
    if (i >= n) return;
    long long row = i / VEXT;
    float p = pmix[row];
    Pw[i] = p * Pgen[i] + (1.f - p) * Pcopy[i];
}

// =====================================================================
extern "C" void kernel_launch(void* const* d_in, const int* in_sizes, int n_in,
                              void* d_out, int out_size, void* d_ws, size_t ws_size,
                              hipStream_t stream) {
    (void)in_sizes; (void)n_in; (void)out_size; (void)ws_size;

    const int*   trg_ids  = (const int*)  d_in[0];
    const float* emofused = (const float*)d_in[1];
    const int*   src_ids  = (const int*)  d_in[2];
    const float* tok_emb  = (const float*)d_in[4];
    const float* pos_emb  = (const float*)d_in[5];
    const float* sa_in_w  = (const float*)d_in[6];
    const float* sa_in_b  = (const float*)d_in[7];
    const float* sa_out_w = (const float*)d_in[8];
    const float* sa_out_b = (const float*)d_in[9];
    const float* ca_in_w  = (const float*)d_in[10];
    const float* ca_in_b  = (const float*)d_in[11];
    const float* ca_out_w = (const float*)d_in[12];
    const float* ca_out_b = (const float*)d_in[13];
    const float* w1       = (const float*)d_in[14];
    const float* b1       = (const float*)d_in[15];
    const float* w2       = (const float*)d_in[16];
    const float* b2       = (const float*)d_in[17];
    const float* n1_g     = (const float*)d_in[18];
    const float* n1_b     = (const float*)d_in[19];
    const float* n2_g     = (const float*)d_in[20];
    const float* n2_b     = (const float*)d_in[21];
    const float* n3_g     = (const float*)d_in[22];
    const float* n3_b     = (const float*)d_in[23];
    const float* vp_w     = (const float*)d_in[24];
    const float* vp_b     = (const float*)d_in[25];
    const float* pg_w     = (const float*)d_in[26];
    const float* pg_b     = (const float*)d_in[27];

    // ---- workspace bump allocator (256B aligned regions) ----
    char* wsb = (char*)d_ws;
    auto alloc = [&](size_t bytes) -> void* {
        void* p = (void*)wsb;
        wsb += (bytes + 255) & ~(size_t)255;
        return p;
    };
    float*  emb      = (float*) alloc((size_t)BTD * 4);
    float*  x1       = (float*) alloc((size_t)BTD * 4);
    float*  x2       = (float*) alloc((size_t)BTD * 4);
    float*  x3       = (float*) alloc((size_t)BTD * 4);
    float*  probs    = (float*) alloc((size_t)2 * BTD * 4);      // B*H*T*LE
    __bf16* h_bf     = (__bf16*)alloc((size_t)BTD * 2);
    __bf16* qkv_bf   = (__bf16*)alloc((size_t)3 * BTD * 2);
    __bf16* sactx_bf = (__bf16*)alloc((size_t)BTD * 2);
    __bf16* qc_bf    = (__bf16*)alloc((size_t)BTD * 2);
    __bf16* kvc_bf   = (__bf16*)alloc((size_t)2 * BTD * 2);
    __bf16* cactx_bf = (__bf16*)alloc((size_t)BTD * 2);
    __bf16* ff_bf    = (__bf16*)alloc((size_t)4 * BTD * 2);      // B*T*DFF
    __bf16* hid_bf   = (__bf16*)alloc((size_t)BTD * 2);
    __bf16* aavg_bf  = (__bf16*)alloc((size_t)BB * TT * LEE * 2);
    __bf16* emo_bf   = (__bf16*)alloc((size_t)BTD * 2);          // B*LE*D
    __bf16* sa_in_wb = (__bf16*)alloc((size_t)3 * DD * DD * 2);
    __bf16* sa_out_wb= (__bf16*)alloc((size_t)DD * DD * 2);
    __bf16* ca_in_wb = (__bf16*)alloc((size_t)3 * DD * DD * 2);
    __bf16* ca_out_wb= (__bf16*)alloc((size_t)DD * DD * 2);
    __bf16* w1_b     = (__bf16*)alloc((size_t)DFFN * DD * 2);
    __bf16* w2_b     = (__bf16*)alloc((size_t)DD * DFFN * 2);
    __bf16* vp_wb    = (__bf16*)alloc((size_t)VV * DD * 2);

    float* out = (float*)d_out;
    const long long PVEXT = (long long)BT * VEXT;
    float* o_Pw      = out;
    float* o_Pgen    = out + PVEXT;
    float* o_Pcopy   = out + 2 * PVEXT;
    float* o_pmix    = out + 3 * PVEXT;
    float* o_hidden  = o_pmix + BT;
    float* o_attnavg = o_hidden + BTD;
    float* o_ctx     = o_attnavg + (long long)BB * TT * LEE;

    auto cvt = [&](const float* s, __bf16* d, long long n) {
        cvt_bf16_kernel<<<(unsigned)((n + 255) / 256), 256, 0, stream>>>(s, d, n);
    };

    // 0) one-time (per launch) bf16 conversion of weights + encoder memory
    cvt(sa_in_w,  sa_in_wb,  (long long)3 * DD * DD);
    cvt(sa_out_w, sa_out_wb, (long long)DD * DD);
    cvt(ca_in_w,  ca_in_wb,  (long long)3 * DD * DD);
    cvt(ca_out_w, ca_out_wb, (long long)DD * DD);
    cvt(w1,       w1_b,      (long long)DFFN * DD);
    cvt(w2,       w2_b,      (long long)DD * DFFN);
    cvt(vp_w,     vp_wb,     (long long)VV * DD);
    cvt(emofused, emo_bf,    (long long)BTD);

    // 1) embedding + positional
    embed_kernel<<<(BTD + 255) / 256, 256, 0, stream>>>(trg_ids, tok_emb, pos_emb, emb, x1);

    // 2) LN1 -> bf16
    ln_kernel<<<BT, 128, 0, stream>>>(x1, n1_g, n1_b, h_bf);

    // 3) self QKV -> qkv_bf (ldc=1536)
    gemm_wmma_bf16<<<dim3(1536 / 64, BT / 64, 1), 128, 0, stream>>>(
        h_bf, sa_in_wb, sa_in_b, nullptr, nullptr, qkv_bf,
        BT, 3 * DD, DD, DD, DD, 3 * DD, 0, 0, 0, 0, 0, 0);

    // 4) causal self-attention -> sactx_bf
    self_attn_kernel<<<BB * HH, 128, 0, stream>>>(qkv_bf, sactx_bf);

    // 5) out-proj + residual(x1) -> x2 (fp32)
    gemm_wmma_bf16<<<dim3(DD / 64, BT / 64, 1), 128, 0, stream>>>(
        sactx_bf, sa_out_wb, sa_out_b, x1, x2, nullptr,
        BT, DD, DD, DD, DD, DD, 0, 0, 0, 0, 0, 0);

    // 6) LN2 -> bf16
    ln_kernel<<<BT, 128, 0, stream>>>(x2, n2_g, n2_b, h_bf);

    // 7) cross Q: W = ca_in_w rows [0,512) -> qc_bf
    gemm_wmma_bf16<<<dim3(DD / 64, BT / 64, 1), 128, 0, stream>>>(
        h_bf, ca_in_wb, ca_in_b, nullptr, nullptr, qc_bf,
        BT, DD, DD, DD, DD, DD, 0, 0, 0, 0, 0, 0);

    // 8) cross K,V from emofused: W = ca_in_w rows [512,1536) -> kvc_bf
    gemm_wmma_bf16<<<dim3(1024 / 64, (BB * LEE) / 64, 1), 128, 0, stream>>>(
        emo_bf, ca_in_wb + (long long)DD * DD, ca_in_b + DD, nullptr, nullptr, kvc_bf,
        BB * LEE, 2 * DD, DD, DD, DD, 2 * DD, 0, 0, 0, 0, 0, 0);

    // 9) cross-attention -> probs (fp32) + cactx_bf
    cross_attn_kernel<<<BB * HH, 128, 0, stream>>>(qc_bf, kvc_bf, probs, cactx_bf);

    // 10) cross out-proj + residual(x2) -> x3 (fp32)
    gemm_wmma_bf16<<<dim3(DD / 64, BT / 64, 1), 128, 0, stream>>>(
        cactx_bf, ca_out_wb, ca_out_b, x2, x3, nullptr,
        BT, DD, DD, DD, DD, DD, 0, 0, 0, 0, 0, 0);

    // 11) LN3 -> bf16
    ln_kernel<<<BT, 128, 0, stream>>>(x3, n3_g, n3_b, h_bf);

    // 12) FFN1 + ReLU -> ff_bf
    gemm_wmma_bf16<<<dim3(DFFN / 64, BT / 64, 1), 128, 0, stream>>>(
        h_bf, w1_b, b1, nullptr, nullptr, ff_bf,
        BT, DFFN, DD, DD, DD, DFFN, 0, 0, 0, 0, 0, 1);

    // 13) FFN2 + residual(x3) -> hidden (fp32 d_out + bf16 for vocab GEMM)
    gemm_wmma_bf16<<<dim3(DD / 64, BT / 64, 1), 128, 0, stream>>>(
        ff_bf, w2_b, b2, x3, o_hidden, hid_bf,
        BT, DD, DFFN, DFFN, DFFN, DD, 0, 0, 0, 0, 0, 0);

    // 14) attn_avg = mean over heads (fp32 d_out + bf16 copy)
    avg_probs_kernel<<<(BB * TT * LEE + 255) / 256, 256, 0, stream>>>(probs, o_attnavg, aavg_bf);

    // 15) ctx = attn_avg @ emofused (batched over B, B is K x N row-major)
    gemm_wmma_bf16<<<dim3(DD / 64, TT / 64, BB), 128, 0, stream>>>(
        aavg_bf, emo_bf, nullptr, nullptr, o_ctx, nullptr,
        TT, DD, LEE, LEE, DD, DD,
        (long long)TT * LEE, (long long)LEE * DD, (long long)TT * DD, 0, 1, 0);

    // 16) vocab logits -> Pgen_ext slot (ldc = VEXT)
    gemm_wmma_bf16<<<dim3(VV / 64, BT / 64, 1), 128, 0, stream>>>(
        hid_bf, vp_wb, vp_b, nullptr, o_Pgen, nullptr,
        BT, VV, DD, DD, DD, VEXT, 0, 0, 0, 0, 0, 0);

    // 17) softmax over vocab + zero pad -> Pgen_ext
    softmax_ext_kernel<<<BT, 256, 0, stream>>>(o_Pgen);

    // 18) gate p_mix = sigmoid([hidden, ctx, emb] . pg_w + pg_b)
    pmix_kernel<<<BT, 128, 0, stream>>>(o_hidden, o_ctx, emb, pg_w, pg_b, o_pmix);

    // 19) zero Pcopy, 20) scatter-add attention mass
    zero_kernel<<<(unsigned)((PVEXT + 255) / 256), 256, 0, stream>>>(o_Pcopy, PVEXT);
    pcopy_scatter_kernel<<<(BB * TT * LEE + 255) / 256, 256, 0, stream>>>(o_attnavg, src_ids, o_Pcopy);

    // 21) Pw = p*Pgen_ext + (1-p)*Pcopy
    pw_kernel<<<(unsigned)((PVEXT + 255) / 256), 256, 0, stream>>>(o_pmix, o_Pgen, o_Pcopy, o_Pw, PVEXT);
}
